// GATConv_45595372814934
// MI455X (gfx1250) — matrix operand order repaired and checked by reading, not compiled, not run on hardware
//
#include <hip/hip_runtime.h>
#include <hip/hip_bf16.h>

#define N_NODES 100000
#define DEG     16
#define N_EDGES (N_NODES * DEG)
#define IN_CH   256
#define OUT_CH  64
#define NEG_SLOPE 0.2f

typedef __attribute__((ext_vector_type(16))) __bf16          v16bf;
typedef __attribute__((ext_vector_type(8)))  float           v8f;
typedef __attribute__((ext_vector_type(8)))  unsigned int    v8ui;
typedef __attribute__((ext_vector_type(16))) unsigned short  v16us;
typedef __attribute__((ext_vector_type(4)))  float           f32x4;
typedef __attribute__((ext_vector_type(2)))  float           f32x2;

// Pack two f32 into one dword of bf16 (round-to-nearest via +0x8000 bias,
// then v_perm_b32 grabs the two high halves in one instruction).
__device__ __forceinline__ unsigned int pack_bf16(float lo, float hi) {
    unsigned int ul = __builtin_bit_cast(unsigned int, lo) + 0x8000u;
    unsigned int uh = __builtin_bit_cast(unsigned int, hi) + 0x8000u;
    // result bytes = [ul.b2, ul.b3, uh.b2, uh.b3]  (sel: S1 bytes 0-3, S0 bytes 4-7)
    return __builtin_amdgcn_perm(uh, ul, 0x07060302u);
}

// scalar f32 -> bf16 (RNE), used only in the tiny one-shot weight prep
__device__ __forceinline__ unsigned short f2bf(float f) {
    unsigned int u = __builtin_bit_cast(unsigned int, f);
    u += 0x7FFFu + ((u >> 16) & 1u);
    return (unsigned short)(u >> 16);
}

// ---------------------------------------------------------------------------
// Kernel 1: rearrange weight [256][64] f32 into WMMA B-fragment order (bf16).
// B order used here (32x16 bf16, contiguous K, same as A): lanes 0-15 hold K=0..15 of col N=lane,
// lanes 16-31 hold K=16..31 of col N=lane-16 (2 bf16 per VGPR, K ascending).
// ---------------------------------------------------------------------------
__global__ __launch_bounds__(256) void prep_weight_kernel(
    const float* __restrict__ w, unsigned short* __restrict__ bfrag) {
    int t = blockIdx.x * blockDim.x + threadIdx.x;
    if (t >= 4 * 8 * 32) return;
    int lane  = t & 31;
    int ks    = (t >> 5) & 7;
    int nt    = t >> 8;
    int n     = nt * 16 + (lane & 15);
    int kbase = ks * 32 + (lane >> 4) * 16;
    unsigned short* dp = bfrag + (size_t)t * 16;
#pragma unroll
    for (int kk = 0; kk < 16; ++kk)
        dp[kk] = f2bf(w[(size_t)(kbase + kk) * OUT_CH + n]);
}

// ---------------------------------------------------------------------------
// Kernel 2: support = x @ weight via v_wmma_f32_16x16x32_bf16.
// One wave per 16-row output strip (16 x 64): 8 K-steps x 4 N-tiles.
// Per K-step: load+convert the A fragment, clause-load all four B fragments,
// then fire the 4 WMMAs.
// ---------------------------------------------------------------------------
__global__ __launch_bounds__(256) void gemm_support_kernel(
    const float* __restrict__ x, const unsigned short* __restrict__ bfrag,
    float* __restrict__ support) {
    int wid  = (blockIdx.x * blockDim.x + threadIdx.x) >> 5;
    int lane = threadIdx.x & 31;
    if (wid >= N_NODES / 16) return;          // 6250 strips, exact fit
    int m0   = wid * 16;
    int mrow = lane & 15;
    int hi   = lane >> 4;                     // lane half selects K sub-range

    v8f acc[4] = {};

#pragma unroll
    for (int ks = 0; ks < 8; ++ks) {
        // A fragment: row m0+mrow, K = ks*32 + hi*16 + {0..15}
        const float* ap = x + (size_t)(m0 + mrow) * IN_CH + ks * 32 + hi * 16;
        f32x4 a0 = *(const f32x4*)(ap + 0);
        f32x4 a1 = *(const f32x4*)(ap + 4);
        f32x4 a2 = *(const f32x4*)(ap + 8);
        f32x4 a3 = *(const f32x4*)(ap + 12);

        // Pull all four B fragments first (one vmem clause, 8x b128).
        v16us b0 = *(const v16us*)(bfrag + ((size_t)(0 * 8 + ks) * 32 + lane) * 16);
        v16us b1 = *(const v16us*)(bfrag + ((size_t)(1 * 8 + ks) * 32 + lane) * 16);
        v16us b2 = *(const v16us*)(bfrag + ((size_t)(2 * 8 + ks) * 32 + lane) * 16);
        v16us b3 = *(const v16us*)(bfrag + ((size_t)(3 * 8 + ks) * 32 + lane) * 16);

        v8ui au;
        au[0] = pack_bf16(a0[0], a0[1]);
        au[1] = pack_bf16(a0[2], a0[3]);
        au[2] = pack_bf16(a1[0], a1[1]);
        au[3] = pack_bf16(a1[2], a1[3]);
        au[4] = pack_bf16(a2[0], a2[1]);
        au[5] = pack_bf16(a2[2], a2[3]);
        au[6] = pack_bf16(a3[0], a3[1]);
        au[7] = pack_bf16(a3[2], a3[3]);
        v16bf afrag = __builtin_bit_cast(v16bf, au);

        acc[0] = __builtin_amdgcn_wmma_f32_16x16x32_bf16(
            false, afrag, false, __builtin_bit_cast(v16bf, b0), (short)0, acc[0], false, false);
        acc[1] = __builtin_amdgcn_wmma_f32_16x16x32_bf16(
            false, afrag, false, __builtin_bit_cast(v16bf, b1), (short)0, acc[1], false, false);
        acc[2] = __builtin_amdgcn_wmma_f32_16x16x32_bf16(
            false, afrag, false, __builtin_bit_cast(v16bf, b2), (short)0, acc[2], false, false);
        acc[3] = __builtin_amdgcn_wmma_f32_16x16x32_bf16(
            false, afrag, false, __builtin_bit_cast(v16bf, b3), (short)0, acc[3], false, false);
    }

    // C/D layout: lanes 0-15 -> N=lane, M=r; lanes 16-31 -> N=lane-16, M=r+8
#pragma unroll
    for (int nt = 0; nt < 4; ++nt)
#pragma unroll
        for (int r = 0; r < 8; ++r)
            support[(size_t)(m0 + hi * 8 + r) * OUT_CH + nt * 16 + (lane & 15)] =
                acc[nt][r];
}

// ---------------------------------------------------------------------------
// Kernel 3: s_src[n] = support[n,:] . att[0:64], s_dst[n] = . att[64:128]
// One wave per node; lane reads a float2, shfl_down reduce.
// ---------------------------------------------------------------------------
__global__ __launch_bounds__(256) void scores_kernel(
    const float* __restrict__ support, const float* __restrict__ att,
    float* __restrict__ s_src, float* __restrict__ s_dst) {
    int wid  = (blockIdx.x * blockDim.x + threadIdx.x) >> 5;
    int lane = threadIdx.x & 31;
    if (wid >= N_NODES) return;
    f32x2 v  = *(const f32x2*)(support + (size_t)wid * OUT_CH + lane * 2);
    f32x2 as = *(const f32x2*)(att + lane * 2);
    f32x2 ad = *(const f32x2*)(att + OUT_CH + lane * 2);
    float ps = v[0] * as[0] + v[1] * as[1];
    float pd = v[0] * ad[0] + v[1] * ad[1];
#pragma unroll
    for (int off = 16; off; off >>= 1) {
        ps += __shfl_down(ps, off);
        pd += __shfl_down(pd, off);
    }
    if (lane == 0) { s_src[wid] = ps; s_dst[wid] = pd; }
}

// ---------------------------------------------------------------------------
// Kernel 4: edge aggregation. Edges are grouped 16-per-source-node, so one
// wave per node handles the whole segment, no atomics. Lanes 0-15 compute
// the 16 edge weights; the MAC loop broadcasts (w_k, dst_k) via v_readlane
// (constant lane index) so the weight is an SGPR operand and the gather
// becomes a scalar-base global_load_b64 per k — no DS traffic in the loop.
// ---------------------------------------------------------------------------
__global__ __launch_bounds__(256) void aggregate_kernel(
    const float* __restrict__ support, const int* __restrict__ dst,
    const float* __restrict__ adj, const float* __restrict__ s_src,
    const float* __restrict__ s_dst, float* __restrict__ out) {
    int wid  = (blockIdx.x * blockDim.x + threadIdx.x) >> 5;
    int lane = threadIdx.x & 31;
    if (wid >= N_NODES) return;
    int n = wid;
    bool valid = lane < 16;
    size_t e = (size_t)n * DEG + (lane & 15);   // lanes 16-31 mirror 0-15

    int   d = dst[e];
    float a = valid ? adj[e] : 0.0f;

    if (valid)
        __builtin_prefetch(support + (size_t)d * OUT_CH, 0, 0);

    // deg[n] = sum of adj over the node's 16 edges (butterfly -> all lanes)
    float deg = a;
#pragma unroll
    for (int off = 16; off; off >>= 1) deg += __shfl_xor(deg, off);

    float sc = s_src[n] + s_dst[d];
    float lr = sc > 0.0f ? sc : sc * NEG_SLOPE;
    float w  = __expf(lr) / deg;

    unsigned int wbits = __builtin_bit_cast(unsigned int, w);

    f32x2 acc = {0.0f, 0.0f};
#pragma unroll
    for (int k = 0; k < DEG; ++k) {
        float wk = __builtin_bit_cast(float,
            (unsigned int)__builtin_amdgcn_readlane((int)wbits, k));
        int dk = __builtin_amdgcn_readlane(d, k);
        f32x2 v = *(const f32x2*)(support + (size_t)dk * OUT_CH + lane * 2);
        acc[0] += wk * v[0];
        acc[1] += wk * v[1];
    }
    *(f32x2*)(out + (size_t)n * OUT_CH + lane * 2) = acc;
}

// ---------------------------------------------------------------------------
// Workspace layout:
//   [0, 25.6MB)            support (100000 x 64 f32)
//   [+0.4MB)               s_src
//   [+0.4MB)               s_dst
//   [+32KB, 32B-aligned)   bf16 B-fragment table
// ---------------------------------------------------------------------------
extern "C" void kernel_launch(void* const* d_in, const int* in_sizes, int n_in,
                              void* d_out, int out_size, void* d_ws, size_t ws_size,
                              hipStream_t stream) {
    const float* x          = (const float*)d_in[0];
    const int*   edge_index = (const int*)d_in[1];
    const float* adj        = (const float*)d_in[2];
    const float* weight     = (const float*)d_in[3];
    const float* attention  = (const float*)d_in[4];
    const int*   dst        = edge_index + (size_t)N_EDGES;  // row 1

    char* ws = (char*)d_ws;
    float*          support = (float*)ws;
    float*          s_src   = (float*)(ws + (size_t)N_NODES * OUT_CH * 4);
    float*          s_dst   = (float*)(ws + (size_t)N_NODES * OUT_CH * 4 + (size_t)N_NODES * 4);
    unsigned short* bfrag   = (unsigned short*)(ws + (size_t)N_NODES * OUT_CH * 4 + (size_t)N_NODES * 8);

    prep_weight_kernel<<<4, 256, 0, stream>>>(weight, bfrag);

    int gemm_waves  = N_NODES / 16;                      // 6250
    int gemm_blocks = (gemm_waves * 32 + 255) / 256;     // 782
    gemm_support_kernel<<<gemm_blocks, 256, 0, stream>>>(x, bfrag, support);

    int node_blocks = (N_NODES * 32 + 255) / 256;        // 12500
    scores_kernel<<<node_blocks, 256, 0, stream>>>(support, attention, s_src, s_dst);
    aggregate_kernel<<<node_blocks, 256, 0, stream>>>(support, dst, adj, s_src, s_dst,
                                                      (float*)d_out);
}
